// LLMJEmodel_87643102642627
// MI455X (gfx1250) — matrix operand order repaired and checked by reading, not compiled, not run on hardware
//
#include <hip/hip_runtime.h>
#include <hip/hip_bf16.h>
#include <math.h>

// ---------------------------------------------------------------------------
// Types for CDNA5 WMMA
// ---------------------------------------------------------------------------
typedef __attribute__((ext_vector_type(16))) _Float16 v16h;
typedef __attribute__((ext_vector_type(8)))  _Float16 h8;
typedef __attribute__((ext_vector_type(8)))  float    v8f;

// ---------------------------------------------------------------------------
// Generic batched GEMM on the matrix engine, f16 operands / f32 accumulate.
//   C[bz] (M,N) = alpha * A[bz](M,K) @ B' (+ bias[n])   (f32 out, optional)
//   Ch    same result stored as f16                      (optional)
//   TRANSB=1 : B stored (N,K) row-major (weight layout, computes A @ W^T)
//   TRANSB=0 : B stored (K,N) row-major (computes A @ B)
// Block = 128 threads = 4 waves; each wave owns a 32x32 C tile (2x2 WMMA
// tiles): 2 A frags x 2 B frags -> 4 back-to-back WMMAs per 32-wide K step.
// Operands are f16 in memory, so fragment loads are bare global_load_b128
// with no convert chain before the WMMAs (half the bytes of the f32 path).
// Requires M%32==0, N%128==0, K%32==0 (true for every GEMM here) so EXEC is
// all-ones everywhere (WMMA requirement) and no bounds checks are needed.
// ---------------------------------------------------------------------------
template <int TRANSB>
__global__ void wmma_gemm_kernel(const _Float16* __restrict__ A,
                                 const _Float16* __restrict__ Bp,
                                 const float* __restrict__ bias,
                                 float* __restrict__ C,
                                 _Float16* __restrict__ Ch,
                                 int M, int N, int K,
                                 long sA, long sB, long sC,
                                 float alpha)
{
    const int lane = threadIdx.x & 31;
    const int wave = threadIdx.x >> 5;
    const int half = lane >> 4;     // which 16-lane half of the wave
    const int lrow = lane & 15;     // A-row / B-column within a 16x16 tile
    const int bz   = blockIdx.z;
    const int m0   = blockIdx.y * 32;
    const int n0   = blockIdx.x * 128 + wave * 32;

    const _Float16* Ab = A  + (long)bz * sA;
    const _Float16* Bb = Bp + (long)bz * sB;

    // Per-lane base pointers, advanced by 32 elements (or 32 rows) per step.
    const _Float16* arow0 = Ab + (long)(m0 + lrow) * K + half * 16;
    const _Float16* arow1 = arow0 + (long)16 * K;
    const _Float16* brow0;
    const _Float16* brow1;
    long bstep;
    if (TRANSB) {
        brow0 = Bb + (long)(n0 + lrow) * K + half * 16;
        brow1 = brow0 + (long)16 * K;
        bstep = 32;
    } else {
        brow0 = Bb + (long)(half * 16) * N + (n0 + lrow);
        brow1 = brow0 + 16;
        bstep = (long)32 * N;
    }

    v8f acc00 = {}, acc01 = {}, acc10 = {}, acc11 = {};
    for (int k0 = 0; k0 < K; k0 += 32) {
        // A fragment: element j -> k = half*16 + j (contiguous, matches B):
        // two contiguous 16B vectors.
        const h8 a0lo = *(const h8*)(arow0);
        const h8 a0hi = *(const h8*)(arow0 + 8);
        const h8 a1lo = *(const h8*)(arow1);
        const h8 a1hi = *(const h8*)(arow1 + 8);
        const v16h a0 = __builtin_shufflevector(a0lo, a0hi, 0,1,2,3,4,5,6,7,8,9,10,11,12,13,14,15);
        const v16h a1 = __builtin_shufflevector(a1lo, a1hi, 0,1,2,3,4,5,6,7,8,9,10,11,12,13,14,15);

        // B fragment: lane holds one column; element j -> k = half*16 + j.
        v16h b0, b1;
        if (TRANSB) {
            const h8 b0lo = *(const h8*)(brow0);
            const h8 b0hi = *(const h8*)(brow0 + 8);
            const h8 b1lo = *(const h8*)(brow1);
            const h8 b1hi = *(const h8*)(brow1 + 8);
            b0 = __builtin_shufflevector(b0lo, b0hi, 0,1,2,3,4,5,6,7,8,9,10,11,12,13,14,15);
            b1 = __builtin_shufflevector(b1lo, b1hi, 0,1,2,3,4,5,6,7,8,9,10,11,12,13,14,15);
        } else {
#pragma unroll
            for (int j = 0; j < 16; ++j) {
                b0[j] = brow0[(long)j * N];
                b1[j] = brow1[(long)j * N];
            }
        }
        acc00 = __builtin_amdgcn_wmma_f32_16x16x32_f16(false, a0, false, b0,
                                                       (short)0, acc00, false, false);
        acc01 = __builtin_amdgcn_wmma_f32_16x16x32_f16(false, a0, false, b1,
                                                       (short)0, acc01, false, false);
        acc10 = __builtin_amdgcn_wmma_f32_16x16x32_f16(false, a1, false, b0,
                                                       (short)0, acc10, false, false);
        acc11 = __builtin_amdgcn_wmma_f32_16x16x32_f16(false, a1, false, b1,
                                                       (short)0, acc11, false, false);
        arow0 += 32; arow1 += 32;
        brow0 += bstep; brow1 += bstep;
    }

    // C/D layout: VGPR v -> m = half*8 + v within the tile, column = lrow.
    const int nA = n0 + lrow;
    const int nB = n0 + 16 + lrow;
    const float bvA = bias ? bias[nA] : 0.0f;
    const float bvB = bias ? bias[nB] : 0.0f;
    if (C) {
        float* Cb = C + (long)bz * sC;
#pragma unroll
        for (int v = 0; v < 8; ++v) {
            const int mA = m0 + half * 8 + v;
            const int mB = mA + 16;
            Cb[(long)mA * N + nA] = acc00[v] * alpha + bvA;
            Cb[(long)mA * N + nB] = acc01[v] * alpha + bvB;
            Cb[(long)mB * N + nA] = acc10[v] * alpha + bvA;
            Cb[(long)mB * N + nB] = acc11[v] * alpha + bvB;
        }
    }
    if (Ch) {
        _Float16* Hb = Ch + (long)bz * sC;
#pragma unroll
        for (int v = 0; v < 8; ++v) {
            const int mA = m0 + half * 8 + v;
            const int mB = mA + 16;
            Hb[(long)mA * N + nA] = (_Float16)(acc00[v] * alpha + bvA);
            Hb[(long)mA * N + nB] = (_Float16)(acc01[v] * alpha + bvB);
            Hb[(long)mB * N + nA] = (_Float16)(acc10[v] * alpha + bvA);
            Hb[(long)mB * N + nB] = (_Float16)(acc11[v] * alpha + bvB);
        }
    }
}

// ---------------------------------------------------------------------------
// Elementwise f32 -> f16 conversion (one-shot operand preparation).
// ---------------------------------------------------------------------------
__global__ void cvt_f16_kernel(const float* __restrict__ x, _Float16* __restrict__ y, long n)
{
    const long i = blockIdx.x * (long)blockDim.x + threadIdx.x;
    if (i < n) y[i] = (_Float16)x[i];
}

// ---------------------------------------------------------------------------
// GRU scan (one direction). One block per batch element, h kept in LDS.
// xg = precomputed X@Wi^T + bi, shape (B,S,3H). Wh (3H,H), bh (3H).
// Writes h_t into out[(b*S+t)*outStride + outOff + u].
// ---------------------------------------------------------------------------
template <int HID>
__global__ void gru_scan_kernel(const float* __restrict__ xg,
                                const float* __restrict__ Wh,
                                const float* __restrict__ bh,
                                float* __restrict__ out,
                                int S, int outStride, int outOff, int reverse)
{
    __shared__ float h[HID];
    const int u = threadIdx.x;
    const int b = blockIdx.x;
    h[u] = 0.0f;
    __syncthreads();

    const float bhr = bh[u], bhz = bh[HID + u], bhn = bh[2 * HID + u];
    const float* Wr = Wh + (long)u * HID;
    const float* Wz = Wh + (long)(HID + u) * HID;
    const float* Wn = Wh + (long)(2 * HID + u) * HID;

    for (int step = 0; step < S; ++step) {
        const int t = reverse ? (S - 1 - step) : step;
        float ar = bhr, az = bhz, an = bhn;
        for (int k = 0; k < HID; ++k) {
            const float hk = h[k];
            ar += hk * Wr[k];
            az += hk * Wz[k];
            an += hk * Wn[k];
        }
        const float* xt = xg + ((long)b * S + t) * (3 * HID);
        const float r = 1.0f / (1.0f + __expf(-(xt[u] + ar)));
        const float z = 1.0f / (1.0f + __expf(-(xt[HID + u] + az)));
        const float n = tanhf(xt[2 * HID + u] + r * an);
        const float hold = h[u];
        const float hnew = (1.0f - z) * n + z * hold;
        __syncthreads();
        h[u] = hnew;
        __syncthreads();
        out[((long)b * S + t) * outStride + outOff + u] = hnew;
    }
}

// ---------------------------------------------------------------------------
// Row softmax: reads f32 scores, stores f16 probabilities (next GEMM operand).
// X is scratch (exp values staged in place). L columns, 256 threads per row.
// ---------------------------------------------------------------------------
__global__ void softmax_rows_kernel(float* __restrict__ X, _Float16* __restrict__ Y, int L)
{
    float* x = X + (long)blockIdx.x * L;
    _Float16* y = Y + (long)blockIdx.x * L;
    __shared__ float red[256];
    const int tid = threadIdx.x;

    float mx = -1e30f;
    for (int j = tid; j < L; j += 256) mx = fmaxf(mx, x[j]);
    red[tid] = mx; __syncthreads();
    for (int w = 128; w > 0; w >>= 1) {
        if (tid < w) red[tid] = fmaxf(red[tid], red[tid + w]);
        __syncthreads();
    }
    mx = red[0]; __syncthreads();

    float sum = 0.0f;
    for (int j = tid; j < L; j += 256) {
        const float e = __expf(x[j] - mx);
        x[j] = e; sum += e;
    }
    red[tid] = sum; __syncthreads();
    for (int w = 128; w > 0; w >>= 1) {
        if (tid < w) red[tid] += red[tid + w];
        __syncthreads();
    }
    const float inv = 1.0f / red[0];
    for (int j = tid; j < L; j += 256) y[j] = (_Float16)(x[j] * inv);
}

// ---------------------------------------------------------------------------
// Narrow output projection (2 outputs): out[m,o] = bias[o] + X[m,:]·W[o,:]
// ---------------------------------------------------------------------------
__global__ void proj2_kernel(const float* __restrict__ X, const float* __restrict__ W,
                             const float* __restrict__ bias, float* __restrict__ out,
                             long M, int K)
{
    const long idx = blockIdx.x * (long)blockDim.x + threadIdx.x;
    if (idx >= M * 2) return;
    const long m = idx >> 1;
    const int  o = (int)(idx & 1);
    const float* x = X + m * K;
    const float* w = W + (long)o * K;
    float acc = bias[o];
    for (int k = 0; k < K; ++k) acc += x[k] * w[k];
    out[idx] = acc;
}

// ---------------------------------------------------------------------------
// Span-mean feature: feat[b,n,d] = mean_{t in [start, end]} pre[b,t,d]
// (reference: (csum[end+1]-csum[start]) / (end+1-start))
// ---------------------------------------------------------------------------
__global__ void feat_kernel(const float* __restrict__ pre, const int* __restrict__ ss,
                            const int* __restrict__ se, float* __restrict__ feat)
{
    const int bn = blockIdx.x;            // b*4 + n
    const int b  = bn >> 2;
    const int s0 = ss[bn];
    const int e  = se[bn];
    const float inv = 1.0f / (float)(e + 1 - s0);
    for (int d = threadIdx.x; d < 768; d += blockDim.x) {
        float acc = 0.0f;
        for (int t = s0; t <= e; ++t) acc += pre[((long)b * 512 + t) * 768 + d];
        feat[(long)bn * 768 + d] = acc * inv;
    }
}

// ---------------------------------------------------------------------------
// RE2[b,s,d] = sum_n mask(b,n,s) * feat[b,n,d], mask = start<=s<end
// ---------------------------------------------------------------------------
__global__ void re2_kernel(const float* __restrict__ feat, const int* __restrict__ ss,
                           const int* __restrict__ se, float* __restrict__ RE2)
{
    const long bs = blockIdx.x;           // b*512 + s
    const int b = (int)(bs >> 9);
    const int s = (int)(bs & 511);
    for (int d = threadIdx.x; d < 768; d += blockDim.x) {
        float acc = 0.0f;
#pragma unroll
        for (int n = 0; n < 4; ++n) {
            if (s >= ss[b * 4 + n] && s < se[b * 4 + n]) acc += feat[(long)(b * 4 + n) * 768 + d];
        }
        RE2[bs * 768 + d] = acc;
    }
}

// ---------------------------------------------------------------------------
// REin[(b,n),s,:] = concat(ctx[b,s,:768], feat[b,n,:768], RE2[b,s,:768]),
// written directly in f16 (next GEMM's A operand).
// ---------------------------------------------------------------------------
__global__ void rein_kernel(const float* __restrict__ ctx, const float* __restrict__ feat,
                            const float* __restrict__ RE2, _Float16* __restrict__ REin)
{
    const long bns = blockIdx.x;          // (b*4+n)*512 + s
    const int bn = (int)(bns >> 9);
    const int s  = (int)(bns & 511);
    const int b  = bn >> 2;
    _Float16* o = REin + bns * 2304;
    const float* c = ctx + ((long)b * 512 + s) * 768;
    const float* f = feat + (long)bn * 768;
    const float* r = RE2 + ((long)b * 512 + s) * 768;
    for (int d = threadIdx.x; d < 768; d += blockDim.x) {
        o[d]        = (_Float16)c[d];
        o[768 + d]  = (_Float16)f[d];
        o[1536 + d] = (_Float16)r[d];
    }
}

// ---------------------------------------------------------------------------
// Host-side launch helpers
// ---------------------------------------------------------------------------
static inline void launch_gemm(const _Float16* A, const _Float16* B, const float* bias,
                               float* C, _Float16* Ch,
                               int M, int N, int K, long sA, long sB, long sC,
                               int batch, int transB, float alpha, hipStream_t st)
{
    dim3 grid(N / 128, M / 32, batch);
    if (transB)
        wmma_gemm_kernel<1><<<grid, 128, 0, st>>>(A, B, bias, C, Ch, M, N, K, sA, sB, sC, alpha);
    else
        wmma_gemm_kernel<0><<<grid, 128, 0, st>>>(A, B, bias, C, Ch, M, N, K, sA, sB, sC, alpha);
}

static inline void launch_cvt(const float* x, _Float16* y, long n, hipStream_t st)
{
    cvt_f16_kernel<<<(unsigned)((n + 255) / 256), 256, 0, st>>>(x, y, n);
}

extern "C" void kernel_launch(void* const* d_in, const int* in_sizes, int n_in,
                              void* d_out, int out_size, void* d_ws, size_t ws_size,
                              hipStream_t stream)
{
    (void)in_sizes; (void)n_in; (void)out_size; (void)ws_size;
    const int B = 8, S = 512, D = 768, NS = 4, H1 = 512, H2 = 384;
    const int BS = B * S;                 // 4096
    const int BNS_S = B * NS * S;         // 16384

    // ----- inputs (setup_inputs() dict order, params flattened in order) ----
    const float* pre = (const float*)d_in[0];
    const int*   ss  = (const int*)d_in[1];
    const int*   se  = (const int*)d_in[2];
    const float* g1_Wi_f = (const float*)d_in[3];
    const float* g1_Wh_f = (const float*)d_in[4];
    const float* g1_bi_f = (const float*)d_in[5];
    const float* g1_bh_f = (const float*)d_in[6];
    const float* g1_Wi_b = (const float*)d_in[7];
    const float* g1_Wh_b = (const float*)d_in[8];
    const float* g1_bi_b = (const float*)d_in[9];
    const float* g1_bh_b = (const float*)d_in[10];
    const float* g2_Wi_f = (const float*)d_in[11];
    const float* g2_Wh_f = (const float*)d_in[12];
    const float* g2_bi_f = (const float*)d_in[13];
    const float* g2_bh_f = (const float*)d_in[14];
    const float* g2_Wi_b = (const float*)d_in[15];
    const float* g2_Wh_b = (const float*)d_in[16];
    const float* g2_bi_b = (const float*)d_in[17];
    const float* g2_bh_b = (const float*)d_in[18];
    const float* W_Q = (const float*)d_in[19];
    const float* W_K = (const float*)d_in[20];
    const float* W_V = (const float*)d_in[21];
    const float* RE_W_Q = (const float*)d_in[22];
    const float* RE_W_K = (const float*)d_in[23];
    const float* RE_W_V = (const float*)d_in[24];
    const float* fc1h_w = (const float*)d_in[25];
    const float* fc1h_b = (const float*)d_in[26];
    const float* fc1t_w = (const float*)d_in[27];
    const float* fc1t_b = (const float*)d_in[28];
    const float* fc2h_w = (const float*)d_in[29];
    const float* fc2h_b = (const float*)d_in[30];
    const float* fc2t_w = (const float*)d_in[31];
    const float* fc2t_b = (const float*)d_in[32];
    const float* REfc1h_w = (const float*)d_in[33];
    const float* REfc1h_b = (const float*)d_in[34];
    const float* REfc1t_w = (const float*)d_in[35];
    const float* REfc1t_b = (const float*)d_in[36];
    const float* REfc2h_w = (const float*)d_in[37];
    const float* REfc2h_b = (const float*)d_in[38];
    const float* REfc2t_w = (const float*)d_in[39];
    const float* REfc2t_b = (const float*)d_in[40];

    float* out = (float*)d_out;
    float* outHead   = out;                     // (8,512,2)
    float* outTail   = out + 8192;              // (8,512,2)
    float* outREhead = out + 16384;             // (32,512,2)
    float* outREtail = out + 49152;             // (32,512,2)

    // ----- workspace layout (byte cursor; every size keeps 16B alignment) --
    char* base = (char*)d_ws;
    size_t off = 0;
    auto allocf = [&](size_t n) { float* p = (float*)(base + off); off += n * sizeof(float); return p; };
    auto alloch = [&](size_t n) { _Float16* p = (_Float16*)(base + off); off += n * sizeof(_Float16); return p; };

    // f32 buffers
    float* xg1f  = allocf((size_t)BS * 1536);
    float* xg1b  = allocf((size_t)BS * 1536);
    float* g1    = allocf((size_t)BS * 1024);
    float* sc1   = allocf((size_t)B * S * S);
    float* fch   = allocf((size_t)BS * 512);
    float* fct   = allocf((size_t)BS * 512);
    float* ctx   = allocf((size_t)BS * 768);
    float* xg2f  = allocf((size_t)BS * 1152);
    float* xg2b  = allocf((size_t)BS * 1152);
    float* feat  = allocf((size_t)B * NS * 768);
    float* RE2   = allocf((size_t)BS * 768);
    float* REsc  = allocf((size_t)B * NS * S * S);
    float* REfch = allocf((size_t)BNS_S * 512);
    float* REfct = allocf((size_t)BNS_S * 512);

    // f16 operand buffers
    _Float16* h_pre    = alloch((size_t)BS * 768);
    _Float16* h_Wi1f   = alloch((size_t)1536 * 768);
    _Float16* h_Wi1b   = alloch((size_t)1536 * 768);
    _Float16* h_Wi2f   = alloch((size_t)1152 * 768);
    _Float16* h_Wi2b   = alloch((size_t)1152 * 768);
    _Float16* h_WQ     = alloch((size_t)512 * 1024);
    _Float16* h_WK     = alloch((size_t)512 * 1024);
    _Float16* h_WV     = alloch((size_t)512 * 1024);
    _Float16* h_REWQ   = alloch((size_t)768 * 2304);
    _Float16* h_REWK   = alloch((size_t)768 * 2304);
    _Float16* h_REWV   = alloch((size_t)768 * 2304);
    _Float16* h_fc1h   = alloch((size_t)512 * 512);
    _Float16* h_fc1t   = alloch((size_t)512 * 512);
    _Float16* h_REfc1h = alloch((size_t)512 * 768);
    _Float16* h_REfc1t = alloch((size_t)512 * 768);
    _Float16* h_g1     = alloch((size_t)BS * 1024);
    _Float16* h_q1     = alloch((size_t)BS * 512);
    _Float16* h_k1     = alloch((size_t)BS * 512);
    _Float16* h_v1     = alloch((size_t)BS * 512);
    _Float16* h_sc1    = alloch((size_t)B * S * S);
    _Float16* h_att1   = alloch((size_t)BS * 512);
    _Float16* h_REin   = alloch((size_t)BNS_S * 2304);
    _Float16* h_REq    = alloch((size_t)BNS_S * 768);
    _Float16* h_REk    = alloch((size_t)BNS_S * 768);
    _Float16* h_REv    = alloch((size_t)BNS_S * 768);
    _Float16* h_REsc   = alloch((size_t)B * NS * S * S);
    _Float16* h_REatt  = alloch((size_t)BNS_S * 768);

    const float sc_scale1  = 1.0f / sqrtf(512.0f);
    const float sc_scaleRE = 1.0f / sqrtf(768.0f);

    // ===== one-shot operand conversions (f32 -> f16)
    launch_cvt(pre,      h_pre,    (long)BS * 768, stream);
    launch_cvt(g1_Wi_f,  h_Wi1f,   (long)1536 * 768, stream);
    launch_cvt(g1_Wi_b,  h_Wi1b,   (long)1536 * 768, stream);
    launch_cvt(g2_Wi_f,  h_Wi2f,   (long)1152 * 768, stream);
    launch_cvt(g2_Wi_b,  h_Wi2b,   (long)1152 * 768, stream);
    launch_cvt(W_Q,      h_WQ,     (long)512 * 1024, stream);
    launch_cvt(W_K,      h_WK,     (long)512 * 1024, stream);
    launch_cvt(W_V,      h_WV,     (long)512 * 1024, stream);
    launch_cvt(RE_W_Q,   h_REWQ,   (long)768 * 2304, stream);
    launch_cvt(RE_W_K,   h_REWK,   (long)768 * 2304, stream);
    launch_cvt(RE_W_V,   h_REWV,   (long)768 * 2304, stream);
    launch_cvt(fc1h_w,   h_fc1h,   (long)512 * 512, stream);
    launch_cvt(fc1t_w,   h_fc1t,   (long)512 * 512, stream);
    launch_cvt(REfc1h_w, h_REfc1h, (long)512 * 768, stream);
    launch_cvt(REfc1t_w, h_REfc1t, (long)512 * 768, stream);

    // ===== GRU1: input projections (WMMA), then LDS scan per direction
    launch_gemm(h_pre, h_Wi1f, g1_bi_f, xg1f, nullptr, BS, 3 * H1, D, 0, 0, 0, 1, 1, 1.0f, stream);
    launch_gemm(h_pre, h_Wi1b, g1_bi_b, xg1b, nullptr, BS, 3 * H1, D, 0, 0, 0, 1, 1, 1.0f, stream);
    gru_scan_kernel<512><<<B, 512, 0, stream>>>(xg1f, g1_Wh_f, g1_bh_f, g1, S, 1024, 0, 0);
    gru_scan_kernel<512><<<B, 512, 0, stream>>>(xg1b, g1_Wh_b, g1_bh_b, g1, S, 1024, 512, 1);
    launch_cvt(g1, h_g1, (long)BS * 1024, stream);

    // ===== Attention 1 over g1 (B,S,1024); q/k/v produced directly in f16
    launch_gemm(h_g1, h_WQ, nullptr, nullptr, h_q1, BS, 512, 1024, 0, 0, 0, 1, 1, 1.0f, stream);
    launch_gemm(h_g1, h_WK, nullptr, nullptr, h_k1, BS, 512, 1024, 0, 0, 0, 1, 1, 1.0f, stream);
    launch_gemm(h_g1, h_WV, nullptr, nullptr, h_v1, BS, 512, 1024, 0, 0, 0, 1, 1, 1.0f, stream);
    // scores = q @ k^T * scale, per batch (f32 out for softmax)
    launch_gemm(h_q1, h_k1, nullptr, sc1, nullptr, S, S, 512,
                (long)S * 512, (long)S * 512, (long)S * S, B, 1, sc_scale1, stream);
    softmax_rows_kernel<<<B * S, 256, 0, stream>>>(sc1, h_sc1, S);
    // att = dist @ v, per batch (B stored (K,N) -> transB=0); f16 out
    launch_gemm(h_sc1, h_v1, nullptr, nullptr, h_att1, S, 512, S,
                (long)S * S, (long)S * 512, (long)S * 512, B, 0, 1.0f, stream);

    // ===== head / tail projections
    launch_gemm(h_att1, h_fc1h, fc1h_b, fch, nullptr, BS, 512, 512, 0, 0, 0, 1, 1, 1.0f, stream);
    launch_gemm(h_att1, h_fc1t, fc1t_b, fct, nullptr, BS, 512, 512, 0, 0, 0, 1, 1, 1.0f, stream);
    proj2_kernel<<<(BS * 2 + 255) / 256, 256, 0, stream>>>(fch, fc2h_w, fc2h_b, outHead, BS, 512);
    proj2_kernel<<<(BS * 2 + 255) / 256, 256, 0, stream>>>(fct, fc2t_w, fc2t_b, outTail, BS, 512);

    // ===== GRU2 -> ctx (B,S,768)
    launch_gemm(h_pre, h_Wi2f, g2_bi_f, xg2f, nullptr, BS, 3 * H2, D, 0, 0, 0, 1, 1, 1.0f, stream);
    launch_gemm(h_pre, h_Wi2b, g2_bi_b, xg2b, nullptr, BS, 3 * H2, D, 0, 0, 0, 1, 1, 1.0f, stream);
    gru_scan_kernel<384><<<B, 384, 0, stream>>>(xg2f, g2_Wh_f, g2_bh_f, ctx, S, 768, 0, 0);
    gru_scan_kernel<384><<<B, 384, 0, stream>>>(xg2b, g2_Wh_b, g2_bh_b, ctx, S, 768, 384, 1);

    // ===== span features, RE2 scatter, REin assembly (f16)
    feat_kernel<<<B * NS, 256, 0, stream>>>(pre, ss, se, feat);
    re2_kernel<<<B * S, 256, 0, stream>>>(feat, ss, se, RE2);
    rein_kernel<<<B * NS * S, 256, 0, stream>>>(ctx, feat, RE2, h_REin);

    // ===== RE attention (dominant GEMMs: 16384 x 768 x 2304, x3)
    launch_gemm(h_REin, h_REWQ, nullptr, nullptr, h_REq, BNS_S, 768, 2304, 0, 0, 0, 1, 1, 1.0f, stream);
    launch_gemm(h_REin, h_REWK, nullptr, nullptr, h_REk, BNS_S, 768, 2304, 0, 0, 0, 1, 1, 1.0f, stream);
    launch_gemm(h_REin, h_REWV, nullptr, nullptr, h_REv, BNS_S, 768, 2304, 0, 0, 0, 1, 1, 1.0f, stream);
    launch_gemm(h_REq, h_REk, nullptr, REsc, nullptr, S, S, 768,
                (long)S * 768, (long)S * 768, (long)S * S, B * NS, 1, sc_scaleRE, stream);
    softmax_rows_kernel<<<B * NS * S, 256, 0, stream>>>(REsc, h_REsc, S);
    launch_gemm(h_REsc, h_REv, nullptr, nullptr, h_REatt, S, 768, S,
                (long)S * S, (long)S * 768, (long)S * 768, B * NS, 0, 1.0f, stream);

    // ===== RE head / tail
    launch_gemm(h_REatt, h_REfc1h, REfc1h_b, REfch, nullptr, BNS_S, 512, 768, 0, 0, 0, 1, 1, 1.0f, stream);
    launch_gemm(h_REatt, h_REfc1t, REfc1t_b, REfct, nullptr, BNS_S, 512, 768, 0, 0, 0, 1, 1, 1.0f, stream);
    proj2_kernel<<<(BNS_S * 2 + 255) / 256, 256, 0, stream>>>(REfch, REfc2h_w, REfc2h_b, outREhead, BNS_S, 512);
    proj2_kernel<<<(BNS_S * 2 + 255) / 256, 256, 0, stream>>>(REfct, REfc2t_w, REfc2t_b, outREtail, BNS_S, 512);
}
